// PID_SSM_Block_38465727103456
// MI455X (gfx1250) — matrix operand order repaired and checked by reading, not compiled, not run on hardware
//
#include <hip/hip_runtime.h>

// ---------------------------------------------------------------------------
// VMamba-style SSM block for MI455X (gfx1250).
//   - GEMMs: v_wmma_f32_16x16x32_f16, register-blocked MTILE=4 (one wave =
//     64x16 output, B fragment reused 4x) with software-pipelined (double
//     buffered) fragment loads so WMMAs overlap outstanding b128 loads.
//   - gemm_out: LDS-staged tile transpose for fully coalesced float4 stores
//     into the (B,C,L) output layout.
//   - Chunked parallel scan (64 chunks x 64 steps -> 3072 waves) with
//     delta-recompute and gated epilogue fused in (no delta / y buffers).
//   - LDS-tiled transpose+LN prep (coalesced loads AND stores).
//   - Memory-bound op (~20 GFLOP vs ~150 MB traffic @ 23.3 TB/s).
// ---------------------------------------------------------------------------

#define DIMC    192
#define DINNER  384
#define DTRANK  12
#define NSTATE  16
#define BATCH   4
#define LLEN    4096
#define BLTOT   (BATCH * LLEN)
#define CHUNK   64
#define NCHUNK  (LLEN / CHUNK)
#define KDBL    (2 * DINNER)          // 768
#define NDBL    (DTRANK + 2 * NSTATE) // 44
#define NDBLPAD 48                    // padded to 3 x 16
#define MTILE   4

typedef __attribute__((ext_vector_type(16))) _Float16 v16h;
typedef __attribute__((ext_vector_type(8)))  _Float16 v8h;
typedef __attribute__((ext_vector_type(8)))  float    v8f;

__device__ __forceinline__ float sigmoidf_(float x) { return 1.0f / (1.0f + __expf(-x)); }
__device__ __forceinline__ float siluf_(float x)    { return x * sigmoidf_(x); }
__device__ __forceinline__ float softplusf_(float x){ return (x > 20.f) ? x : log1pf(__expf(x)); }

// ---------------------------------------------------------------------------
// Fragment loaders for V_WMMA_F32_16X16X32_F16 (wave32).
// A (16x32, f16): lane m=lane&15, hi=lane>>4; e=0..7 -> k=hi*8+e,
//                 e=8..15 -> k=16+hi*8+(e-8).    => two 16B loads.
// B (32x16, f16): lane n=lane&15, hi=lane>>4; same k mapping as A.
//                 (sourced from row-major W[n][k]) => two 16B loads.
// No guards: every weight buffer is padded to a multiple of 16 rows.
// ---------------------------------------------------------------------------
__device__ __forceinline__ v16h load_fragA(const _Float16* __restrict__ A,
                                           int lda, int k0, int m, int hi) {
    const _Float16* p = A + (size_t)m * lda + k0 + hi * 8;
    v8h lo = *(const v8h*)p;
    v8h hh = *(const v8h*)(p + 16);
    v16h r;
#pragma unroll
    for (int i = 0; i < 8; ++i) { r[i] = lo[i]; r[i + 8] = hh[i]; }
    return r;
}

__device__ __forceinline__ v16h load_fragB(const _Float16* __restrict__ W,
                                           int ldw, int k0, int n, int hi) {
    const _Float16* p = W + (size_t)n * ldw + k0 + hi * 8;
    v8h lo = *(const v8h*)p;
    v8h hh = *(const v8h*)(p + 16);
    v16h r;
#pragma unroll
    for (int i = 0; i < 8; ++i) { r[i] = lo[i]; r[i + 8] = hh[i]; }
    return r;
}

// ---------------------------------------------------------------------------
// Kernel 0: f32 -> f16 weight conversion with optional zero padding
// ---------------------------------------------------------------------------
__global__ void cvt_kernel(const float* __restrict__ s, _Float16* __restrict__ d,
                           int n_src, int n_dst) {
    int i = blockIdx.x * blockDim.x + threadIdx.x;
    if (i < n_dst) d[i] = (i < n_src) ? (_Float16)s[i] : (_Float16)0.f;
}

// ---------------------------------------------------------------------------
// Kernel 1a: per-(b,l) layernorm statistics for K and Q (coalesced reads)
// ---------------------------------------------------------------------------
__global__ void stats_kernel(const float* __restrict__ K, const float* __restrict__ Q,
                             float* __restrict__ kmean, float* __restrict__ krstd,
                             float* __restrict__ qmean, float* __restrict__ qrstd) {
    int idx = blockIdx.x * blockDim.x + threadIdx.x; // b*L + l
    if (idx >= BLTOT) return;
    int b = idx / LLEN, l = idx - b * LLEN;
    const float* Kb = K + (size_t)b * DIMC * LLEN + l;
    const float* Qb = Q + (size_t)b * DIMC * LLEN + l;
    float sk = 0.f, sk2 = 0.f, sq = 0.f, sq2 = 0.f;
    for (int c = 0; c < DIMC; ++c) {
        float kv = Kb[(size_t)c * LLEN];
        float qv = Qb[(size_t)c * LLEN];
        sk += kv; sk2 += kv * kv;
        sq += qv; sq2 += qv * qv;
    }
    const float inv = 1.0f / (float)DIMC;
    float mk = sk * inv, vk = sk2 * inv - mk * mk;
    float mq = sq * inv, vq = sq2 * inv - mq * mq;
    kmean[idx] = mk; krstd[idx] = rsqrtf(vk + 1e-5f);
    qmean[idx] = mq; qrstd[idx] = rsqrtf(vq + 1e-5f);
}

// ---------------------------------------------------------------------------
// Kernel 1b: LDS-tiled 32x32 transpose (B,C,L)->(B,L,C) f16, LN applied on the
// fly for K and Q.  Coalesced global loads AND stores.
// ---------------------------------------------------------------------------
__global__ __launch_bounds__(256)
void prep_t_kernel(const float* __restrict__ x, const float* __restrict__ K,
                   const float* __restrict__ Q,
                   const float* __restrict__ kg, const float* __restrict__ kb,
                   const float* __restrict__ qg, const float* __restrict__ qb,
                   const float* __restrict__ kmean, const float* __restrict__ krstd,
                   const float* __restrict__ qmean, const float* __restrict__ qrstd,
                   _Float16* __restrict__ xf16, _Float16* __restrict__ kf16,
                   _Float16* __restrict__ qf16) {
    __shared__ float tile[32][33];
    int l0 = blockIdx.x * 32;
    int c0 = blockIdx.y * 32;
    int b  = blockIdx.z;
    int tid = threadIdx.x;
    int fast = tid & 31;   // load phase: l ; store phase: c
    int slow = tid >> 5;   // 0..7

#pragma unroll
    for (int which = 0; which < 3; ++which) {
        const float* src = (which == 0) ? x : (which == 1) ? K : Q;
        _Float16* dst = (which == 0) ? xf16 : (which == 1) ? kf16 : qf16;
        __syncthreads();
#pragma unroll
        for (int r = 0; r < 4; ++r) {
            int cc = slow + r * 8;
            tile[cc][fast] = src[((size_t)b * DIMC + c0 + cc) * LLEN + l0 + fast];
        }
        __syncthreads();
#pragma unroll
        for (int r = 0; r < 4; ++r) {
            int ll = slow + r * 8;
            int l = l0 + ll;
            int c = c0 + fast;
            float v = tile[fast][ll];
            if (which == 1) v = (v - kmean[(size_t)b * LLEN + l]) * krstd[(size_t)b * LLEN + l] * kg[c] + kb[c];
            if (which == 2) v = (v - qmean[(size_t)b * LLEN + l]) * qrstd[(size_t)b * LLEN + l] * qg[c] + qb[c];
            dst[((size_t)b * LLEN + l) * DIMC + c] = (_Float16)v;
        }
    }
}

// ---------------------------------------------------------------------------
// Kernel 2: three (B*L,192)@(192,384) GEMMs: x_in, K_proj(silu+b), Q_proj(silu+b)
// One wave per 64x16 tile; double-buffered fragment loads.
// ---------------------------------------------------------------------------
__global__ __launch_bounds__(32)
void gemm_proj_kernel(const _Float16* __restrict__ xf16, const _Float16* __restrict__ kf16,
                      const _Float16* __restrict__ qf16,
                      const _Float16* __restrict__ w_in, const _Float16* __restrict__ w_k,
                      const _Float16* __restrict__ w_q,
                      const float* __restrict__ kbias, const float* __restrict__ qbias,
                      float* __restrict__ x_in, _Float16* __restrict__ kp16,
                      _Float16* __restrict__ qp16) {
    int lane = threadIdx.x;
    int hi = lane >> 4, lo16 = lane & 15;
    int m0 = blockIdx.x * (16 * MTILE);
    int n0 = blockIdx.y * 16;
    int bz = blockIdx.z;
    int b = bz / 3, which = bz % 3;
    const _Float16* A = (which == 0 ? xf16 : which == 1 ? kf16 : qf16)
                        + ((size_t)b * LLEN + m0) * DIMC;
    const _Float16* W = (which == 0 ? w_in : which == 1 ? w_k : w_q);

    v8f acc[MTILE] = {};
    v16h bf = load_fragB(W, DIMC, 0, n0 + lo16, hi);
    v16h af[MTILE];
#pragma unroll
    for (int mt = 0; mt < MTILE; ++mt)
        af[mt] = load_fragA(A + (size_t)mt * 16 * DIMC, DIMC, 0, lo16, hi);

#pragma unroll
    for (int k0 = 0; k0 < DIMC; k0 += 32) {
        v16h bf_n = bf;
        v16h af_n[MTILE];
#pragma unroll
        for (int mt = 0; mt < MTILE; ++mt) af_n[mt] = af[mt];
        if (k0 + 32 < DIMC) {
            bf_n = load_fragB(W, DIMC, k0 + 32, n0 + lo16, hi);
#pragma unroll
            for (int mt = 0; mt < MTILE; ++mt)
                af_n[mt] = load_fragA(A + (size_t)mt * 16 * DIMC, DIMC, k0 + 32, lo16, hi);
        }
#pragma unroll
        for (int mt = 0; mt < MTILE; ++mt)
            acc[mt] = __builtin_amdgcn_wmma_f32_16x16x32_f16(false, af[mt], false, bf,
                                                             (short)0, acc[mt], false, false);
        bf = bf_n;
#pragma unroll
        for (int mt = 0; mt < MTILE; ++mt) af[mt] = af_n[mt];
    }

    int n = n0 + lo16;
#pragma unroll
    for (int mt = 0; mt < MTILE; ++mt) {
#pragma unroll
        for (int r = 0; r < 8; ++r) {
            int m = m0 + mt * 16 + hi * 8 + r;
            size_t o = ((size_t)b * LLEN + m) * DINNER + n;
            float v = acc[mt][r];
            if (which == 0)      x_in[o] = v;
            else if (which == 1) kp16[o] = (_Float16)siluf_(v + kbias[n]);
            else                 qp16[o] = (_Float16)siluf_(v + qbias[n]);
        }
    }
}

// ---------------------------------------------------------------------------
// Kernel 3: causal depthwise conv (k=4) + silu ; f32 and f16 copies
// ---------------------------------------------------------------------------
__global__ __launch_bounds__(DINNER)
void conv_kernel(const float* __restrict__ x_in, const float* __restrict__ cw,
                 const float* __restrict__ cb,
                 float* __restrict__ xc, _Float16* __restrict__ xc16) {
    int d = threadIdx.x;
    int l = blockIdx.x;
    int b = blockIdx.y;
    const float* base = x_in + (size_t)b * LLEN * DINNER + d;
    float acc = cb[d];
#pragma unroll
    for (int j = 0; j < 4; ++j) {
        int ls = l - 3 + j;
        if (ls >= 0) acc += cw[d * 4 + j] * base[(size_t)ls * DINNER];
    }
    float v = siluf_(acc);
    size_t o = ((size_t)b * LLEN + l) * DINNER + d;
    xc[o] = v;
    xc16[o] = (_Float16)v;
}

// ---------------------------------------------------------------------------
// Kernel 4: x_dbl = [xc | Kp] @ dtbc_w^T   (K=768, N=44 zero-padded to 48)
// ---------------------------------------------------------------------------
__global__ __launch_bounds__(32)
void gemm_dbl_kernel(const _Float16* __restrict__ xc16, const _Float16* __restrict__ kp16,
                     const _Float16* __restrict__ w_dtbc,
                     float* __restrict__ dt, float* __restrict__ Bs, float* __restrict__ Cs) {
    int lane = threadIdx.x;
    int hi = lane >> 4, lo16 = lane & 15;
    int m0 = blockIdx.x * (16 * MTILE);
    int n0 = blockIdx.y * 16;
    int b = blockIdx.z;
    const _Float16* A0 = xc16 + ((size_t)b * LLEN + m0) * DINNER;
    const _Float16* A1 = kp16 + ((size_t)b * LLEN + m0) * DINNER;

    v8f acc[MTILE] = {};
    v16h bf = load_fragB(w_dtbc, KDBL, 0, n0 + lo16, hi);
    v16h af[MTILE];
#pragma unroll
    for (int mt = 0; mt < MTILE; ++mt)
        af[mt] = load_fragA(A0 + (size_t)mt * 16 * DINNER, DINNER, 0, lo16, hi);

#pragma unroll
    for (int k0 = 0; k0 < KDBL; k0 += 32) {
        v16h bf_n = bf;
        v16h af_n[MTILE];
#pragma unroll
        for (int mt = 0; mt < MTILE; ++mt) af_n[mt] = af[mt];
        if (k0 + 32 < KDBL) {
            int kn = k0 + 32;
            const _Float16* An = (kn < DINNER) ? A0 : A1;
            int kk = kn & (DINNER - 1);
            bf_n = load_fragB(w_dtbc, KDBL, kn, n0 + lo16, hi);
#pragma unroll
            for (int mt = 0; mt < MTILE; ++mt)
                af_n[mt] = load_fragA(An + (size_t)mt * 16 * DINNER, DINNER, kk, lo16, hi);
        }
#pragma unroll
        for (int mt = 0; mt < MTILE; ++mt)
            acc[mt] = __builtin_amdgcn_wmma_f32_16x16x32_f16(false, af[mt], false, bf,
                                                             (short)0, acc[mt], false, false);
        bf = bf_n;
#pragma unroll
        for (int mt = 0; mt < MTILE; ++mt) af[mt] = af_n[mt];
    }

    int col = n0 + lo16;
#pragma unroll
    for (int mt = 0; mt < MTILE; ++mt) {
#pragma unroll
        for (int r = 0; r < 8; ++r) {
            int m = m0 + mt * 16 + hi * 8 + r;
            size_t row = (size_t)b * LLEN + m;
            float v = acc[mt][r];
            if (col < DTRANK)           dt[row * DTRANK + col]                 = v;
            else if (col < DTRANK + 16) Bs[row * NSTATE + (col - DTRANK)]      = v;
            else if (col < NDBL)        Cs[row * NSTATE + (col - DTRANK - 16)] = v;
        }
    }
}

// ---------------------------------------------------------------------------
// Kernel 5: z = [xc | Qp] @ gate_w^T + gate_b  (K=768, N=384)
// ---------------------------------------------------------------------------
__global__ __launch_bounds__(32)
void gemm_z_kernel(const _Float16* __restrict__ xc16, const _Float16* __restrict__ qp16,
                   const _Float16* __restrict__ w_gate, const float* __restrict__ gbias,
                   float* __restrict__ z) {
    int lane = threadIdx.x;
    int hi = lane >> 4, lo16 = lane & 15;
    int m0 = blockIdx.x * (16 * MTILE);
    int n0 = blockIdx.y * 16;
    int b = blockIdx.z;
    const _Float16* A0 = xc16 + ((size_t)b * LLEN + m0) * DINNER;
    const _Float16* A1 = qp16 + ((size_t)b * LLEN + m0) * DINNER;

    v8f acc[MTILE] = {};
    v16h bf = load_fragB(w_gate, KDBL, 0, n0 + lo16, hi);
    v16h af[MTILE];
#pragma unroll
    for (int mt = 0; mt < MTILE; ++mt)
        af[mt] = load_fragA(A0 + (size_t)mt * 16 * DINNER, DINNER, 0, lo16, hi);

#pragma unroll
    for (int k0 = 0; k0 < KDBL; k0 += 32) {
        v16h bf_n = bf;
        v16h af_n[MTILE];
#pragma unroll
        for (int mt = 0; mt < MTILE; ++mt) af_n[mt] = af[mt];
        if (k0 + 32 < KDBL) {
            int kn = k0 + 32;
            const _Float16* An = (kn < DINNER) ? A0 : A1;
            int kk = kn & (DINNER - 1);
            bf_n = load_fragB(w_gate, KDBL, kn, n0 + lo16, hi);
#pragma unroll
            for (int mt = 0; mt < MTILE; ++mt)
                af_n[mt] = load_fragA(An + (size_t)mt * 16 * DINNER, DINNER, kk, lo16, hi);
        }
#pragma unroll
        for (int mt = 0; mt < MTILE; ++mt)
            acc[mt] = __builtin_amdgcn_wmma_f32_16x16x32_f16(false, af[mt], false, bf,
                                                             (short)0, acc[mt], false, false);
        bf = bf_n;
#pragma unroll
        for (int mt = 0; mt < MTILE; ++mt) af[mt] = af_n[mt];
    }

    int n = n0 + lo16;
#pragma unroll
    for (int mt = 0; mt < MTILE; ++mt) {
#pragma unroll
        for (int r = 0; r < 8; ++r) {
            int m = m0 + mt * 16 + hi * 8 + r;
            z[((size_t)b * LLEN + m) * DINNER + n] = acc[mt][r] + gbias[n];
        }
    }
}

// ---------------------------------------------------------------------------
// delta recompute helper: softplus(dt_row . wdt + 2*dt_b[d])
// ---------------------------------------------------------------------------
__device__ __forceinline__ float delta_of(const float* __restrict__ dtrow,
                                          const float* __restrict__ wdt, float db2) {
    float acc = db2;
#pragma unroll
    for (int r = 0; r < DTRANK; ++r) acc += dtrow[r] * wdt[r];
    return softplusf_(acc);
}

// ---------------------------------------------------------------------------
// Kernel 6: scan pass 1 — per-chunk transition P = exp(a*sum(delta)), q = h
// ---------------------------------------------------------------------------
__global__ __launch_bounds__(DINNER)
void scan1_kernel(const float* __restrict__ dt, const float* __restrict__ dt_w,
                  const float* __restrict__ dt_b,
                  const float* __restrict__ xc, const float* __restrict__ Bs,
                  const float* __restrict__ A_log,
                  float* __restrict__ cP, float* __restrict__ cH) {
    int d = threadIdx.x;
    int s = blockIdx.x;
    int b = blockIdx.y;
    float wdt[DTRANK];
#pragma unroll
    for (int r = 0; r < DTRANK; ++r) wdt[r] = dt_w[d * DTRANK + r];
    float db2 = 2.0f * dt_b[d];
    float a[NSTATE];
#pragma unroll
    for (int n = 0; n < NSTATE; ++n) a[n] = -__expf(A_log[d * NSTATE + n]);
    float h[NSTATE];
#pragma unroll
    for (int n = 0; n < NSTATE; ++n) h[n] = 0.f;
    float sd = 0.f;
    size_t row0 = (size_t)b * LLEN + (size_t)s * CHUNK;
    for (int t = 0; t < CHUNK; ++t) {
        size_t row = row0 + t;
        if (t + 1 < CHUNK) __builtin_prefetch(&xc[(row + 1) * DINNER + d], 0, 0);
        float de = delta_of(dt + row * DTRANK, wdt, db2);
        float u  = xc[row * DINNER + d];
        sd += de;
        float du = de * u;
        const float* Bp = Bs + row * NSTATE;
#pragma unroll
        for (int n = 0; n < NSTATE; ++n)
            h[n] = __expf(de * a[n]) * h[n] + du * Bp[n];
    }
    size_t o = (((size_t)b * NCHUNK + s) * DINNER + d) * NSTATE;
#pragma unroll
    for (int n = 0; n < NSTATE; ++n) {
        cP[o + n] = __expf(a[n] * sd);
        cH[o + n] = h[n];
    }
}

// ---------------------------------------------------------------------------
// Kernel 7: scan pass 2 — sequential combine over 64 chunks (tiny)
// ---------------------------------------------------------------------------
__global__ __launch_bounds__(DINNER)
void scan2_kernel(const float* __restrict__ cP, const float* __restrict__ cH,
                  float* __restrict__ Hin) {
    int d = threadIdx.x;
    int b = blockIdx.x;
    float h[NSTATE];
#pragma unroll
    for (int n = 0; n < NSTATE; ++n) h[n] = 0.f;
    for (int s = 0; s < NCHUNK; ++s) {
        size_t o = (((size_t)b * NCHUNK + s) * DINNER + d) * NSTATE;
#pragma unroll
        for (int n = 0; n < NSTATE; ++n) {
            Hin[o + n] = h[n];
            h[n] = cP[o + n] * h[n] + cH[o + n];
        }
    }
}

// ---------------------------------------------------------------------------
// Kernel 8: scan pass 3 — replay with correct h_in; FUSED gated epilogue:
// yg16 = (y + u*D) * silu(z)  written directly
// ---------------------------------------------------------------------------
__global__ __launch_bounds__(DINNER)
void scan3_kernel(const float* __restrict__ dt, const float* __restrict__ dt_w,
                  const float* __restrict__ dt_b,
                  const float* __restrict__ xc, const float* __restrict__ Bs,
                  const float* __restrict__ Cs, const float* __restrict__ A_log,
                  const float* __restrict__ Hin, const float* __restrict__ z,
                  const float* __restrict__ Dp, _Float16* __restrict__ yg16) {
    int d = threadIdx.x;
    int s = blockIdx.x;
    int b = blockIdx.y;
    float wdt[DTRANK];
#pragma unroll
    for (int r = 0; r < DTRANK; ++r) wdt[r] = dt_w[d * DTRANK + r];
    float db2 = 2.0f * dt_b[d];
    float dpar = Dp[d];
    float a[NSTATE];
#pragma unroll
    for (int n = 0; n < NSTATE; ++n) a[n] = -__expf(A_log[d * NSTATE + n]);
    float h[NSTATE];
    size_t ho = (((size_t)b * NCHUNK + s) * DINNER + d) * NSTATE;
#pragma unroll
    for (int n = 0; n < NSTATE; ++n) h[n] = Hin[ho + n];
    size_t row0 = (size_t)b * LLEN + (size_t)s * CHUNK;
    for (int t = 0; t < CHUNK; ++t) {
        size_t row = row0 + t;
        if (t + 1 < CHUNK) {
            __builtin_prefetch(&xc[(row + 1) * DINNER + d], 0, 0);
            __builtin_prefetch(&z[(row + 1) * DINNER + d], 0, 0);
        }
        float de = delta_of(dt + row * DTRANK, wdt, db2);
        float u  = xc[row * DINNER + d];
        float du = de * u;
        const float* Bp = Bs + row * NSTATE;
        const float* Cp = Cs + row * NSTATE;
        float yv = 0.f;
#pragma unroll
        for (int n = 0; n < NSTATE; ++n) {
            h[n] = __expf(de * a[n]) * h[n] + du * Bp[n];
            yv += h[n] * Cp[n];
        }
        float zz = z[row * DINNER + d];
        yg16[row * DINNER + d] = (_Float16)((yv + u * dpar) * siluf_(zz));
    }
}

// ---------------------------------------------------------------------------
// Kernel 9: out[b,c,l] = yg @ out_w^T  (K=384, N=192).
// LDS-staged 64(l)x16(c) tile -> fully coalesced float4 stores along l.
// Row stride 68 floats: 16B-aligned rows for b128 LDS ops + conflict-free.
// ---------------------------------------------------------------------------
__global__ __launch_bounds__(32)
void gemm_out_kernel(const _Float16* __restrict__ yg16, const _Float16* __restrict__ w_out,
                     float* __restrict__ out) {
    __shared__ float st[16][68];
    int lane = threadIdx.x;
    int hi = lane >> 4, lo16 = lane & 15;
    int m0 = blockIdx.x * (16 * MTILE);   // l
    int n0 = blockIdx.y * 16;             // c
    int b = blockIdx.z;
    const _Float16* A = yg16 + ((size_t)b * LLEN + m0) * DINNER;

    v8f acc[MTILE] = {};
    v16h bf = load_fragB(w_out, DINNER, 0, n0 + lo16, hi);
    v16h af[MTILE];
#pragma unroll
    for (int mt = 0; mt < MTILE; ++mt)
        af[mt] = load_fragA(A + (size_t)mt * 16 * DINNER, DINNER, 0, lo16, hi);

#pragma unroll
    for (int k0 = 0; k0 < DINNER; k0 += 32) {
        v16h bf_n = bf;
        v16h af_n[MTILE];
#pragma unroll
        for (int mt = 0; mt < MTILE; ++mt) af_n[mt] = af[mt];
        if (k0 + 32 < DINNER) {
            bf_n = load_fragB(w_out, DINNER, k0 + 32, n0 + lo16, hi);
#pragma unroll
            for (int mt = 0; mt < MTILE; ++mt)
                af_n[mt] = load_fragA(A + (size_t)mt * 16 * DINNER, DINNER, k0 + 32, lo16, hi);
        }
#pragma unroll
        for (int mt = 0; mt < MTILE; ++mt)
            acc[mt] = __builtin_amdgcn_wmma_f32_16x16x32_f16(false, af[mt], false, bf,
                                                             (short)0, acc[mt], false, false);
        bf = bf_n;
#pragma unroll
        for (int mt = 0; mt < MTILE; ++mt) af[mt] = af_n[mt];
    }

    // stage tile: st[c_local][l_local]
#pragma unroll
    for (int mt = 0; mt < MTILE; ++mt) {
#pragma unroll
        for (int r = 0; r < 8; ++r) {
            st[lo16][mt * 16 + hi * 8 + r] = acc[mt][r];
        }
    }
    __syncthreads();   // single-wave workgroup: lowers to S_NOP; ds waits handle order

    // coalesced write: half-wave per c-row, float4 per lane along l
    int q = lo16;
#pragma unroll
    for (int it = 0; it < 8; ++it) {
        int c = it * 2 + hi;
        float4 v = *(const float4*)&st[c][q * 4];
        *(float4*)&out[((size_t)b * DIMC + n0 + c) * LLEN + m0 + q * 4] = v;
    }
}

// ---------------------------------------------------------------------------
// Host launcher.  Workspace requirement ~168 MB.
// ---------------------------------------------------------------------------
extern "C" void kernel_launch(void* const* d_in, const int* in_sizes, int n_in,
                              void* d_out, int out_size, void* d_ws, size_t ws_size,
                              hipStream_t stream) {
    const float* x      = (const float*)d_in[0];
    const float* K      = (const float*)d_in[1];
    const float* Q      = (const float*)d_in[2];
    const float* in_w   = (const float*)d_in[3];
    const float* conv_w = (const float*)d_in[4];
    const float* conv_b = (const float*)d_in[5];
    const float* k_ln_g = (const float*)d_in[6];
    const float* k_ln_b = (const float*)d_in[7];
    const float* k_w    = (const float*)d_in[8];
    const float* k_b    = (const float*)d_in[9];
    const float* q_ln_g = (const float*)d_in[10];
    const float* q_ln_b = (const float*)d_in[11];
    const float* q_w    = (const float*)d_in[12];
    const float* q_b    = (const float*)d_in[13];
    const float* dtbc_w = (const float*)d_in[14];
    const float* dt_w   = (const float*)d_in[15];
    const float* dt_b   = (const float*)d_in[16];
    const float* gate_w = (const float*)d_in[17];
    const float* gate_b = (const float*)d_in[18];
    const float* A_log  = (const float*)d_in[19];
    const float* D_par  = (const float*)d_in[20];
    const float* out_w  = (const float*)d_in[21];
    float* out = (float*)d_out;

    char* base = (char*)d_ws;
    size_t off = 0;
    auto alloc = [&](size_t bytes) -> void* {
        void* p = base + off;
        off = (off + bytes + 255) & ~(size_t)255;
        return p;
    };

    _Float16* w_in16   = (_Float16*)alloc((size_t)DINNER * DIMC * 2);
    _Float16* w_k16    = (_Float16*)alloc((size_t)DINNER * DIMC * 2);
    _Float16* w_q16    = (_Float16*)alloc((size_t)DINNER * DIMC * 2);
    _Float16* w_dtbc16 = (_Float16*)alloc((size_t)NDBLPAD * KDBL * 2); // zero-padded rows 44..47
    _Float16* w_gate16 = (_Float16*)alloc((size_t)DINNER * KDBL * 2);
    _Float16* w_out16  = (_Float16*)alloc((size_t)DIMC * DINNER * 2);

    float* kmean = (float*)alloc((size_t)BLTOT * 4);
    float* krstd = (float*)alloc((size_t)BLTOT * 4);
    float* qmean = (float*)alloc((size_t)BLTOT * 4);
    float* qrstd = (float*)alloc((size_t)BLTOT * 4);

    _Float16* xf16 = (_Float16*)alloc((size_t)BLTOT * DIMC * 2);
    _Float16* kf16 = (_Float16*)alloc((size_t)BLTOT * DIMC * 2);
    _Float16* qf16 = (_Float16*)alloc((size_t)BLTOT * DIMC * 2);

    float*    x_in = (float*)   alloc((size_t)BLTOT * DINNER * 4);
    _Float16* kp16 = (_Float16*)alloc((size_t)BLTOT * DINNER * 2);
    _Float16* qp16 = (_Float16*)alloc((size_t)BLTOT * DINNER * 2);
    float*    xc   = (float*)   alloc((size_t)BLTOT * DINNER * 4);
    _Float16* xc16 = (_Float16*)alloc((size_t)BLTOT * DINNER * 2);

    float* dtbuf = (float*)alloc((size_t)BLTOT * DTRANK * 4);
    float* Bsbuf = (float*)alloc((size_t)BLTOT * NSTATE * 4);
    float* Csbuf = (float*)alloc((size_t)BLTOT * NSTATE * 4);
    float* zbuf  = (float*)alloc((size_t)BLTOT * DINNER * 4);

    float* cP  = (float*)alloc((size_t)BATCH * NCHUNK * DINNER * NSTATE * 4);
    float* cH  = (float*)alloc((size_t)BATCH * NCHUNK * DINNER * NSTATE * 4);
    float* Hin = (float*)alloc((size_t)BATCH * NCHUNK * DINNER * NSTATE * 4);

    _Float16* yg16 = (_Float16*)alloc((size_t)BLTOT * DINNER * 2);

    // ---- weight conversion (dtbc zero-padded to 48 rows) ----
    auto cvt = [&](const float* s, _Float16* d, int n_src, int n_dst) {
        cvt_kernel<<<(n_dst + 255) / 256, 256, 0, stream>>>(s, d, n_src, n_dst);
    };
    cvt(in_w,   w_in16,   DINNER * DIMC, DINNER * DIMC);
    cvt(k_w,    w_k16,    DINNER * DIMC, DINNER * DIMC);
    cvt(q_w,    w_q16,    DINNER * DIMC, DINNER * DIMC);
    cvt(dtbc_w, w_dtbc16, NDBL * KDBL,   NDBLPAD * KDBL);
    cvt(gate_w, w_gate16, DINNER * KDBL, DINNER * KDBL);
    cvt(out_w,  w_out16,  DIMC * DINNER, DIMC * DINNER);

    // ---- LN stats + LDS-tiled transpose ----
    stats_kernel<<<BLTOT / 256, 256, 0, stream>>>(K, Q, kmean, krstd, qmean, qrstd);
    prep_t_kernel<<<dim3(LLEN / 32, DIMC / 32, BATCH), 256, 0, stream>>>(
        x, K, Q, k_ln_g, k_ln_b, q_ln_g, q_ln_b,
        kmean, krstd, qmean, qrstd, xf16, kf16, qf16);

    // ---- three projection GEMMs ----
    gemm_proj_kernel<<<dim3(LLEN / (16 * MTILE), DINNER / 16, BATCH * 3), 32, 0, stream>>>(
        xf16, kf16, qf16, w_in16, w_k16, w_q16, k_b, q_b, x_in, kp16, qp16);

    // ---- causal depthwise conv + silu ----
    conv_kernel<<<dim3(LLEN, BATCH), DINNER, 0, stream>>>(x_in, conv_w, conv_b, xc, xc16);

    // ---- x_dbl GEMM (dt / B / C split) ----
    gemm_dbl_kernel<<<dim3(LLEN / (16 * MTILE), 3, BATCH), 32, 0, stream>>>(
        xc16, kp16, w_dtbc16, dtbuf, Bsbuf, Csbuf);

    // ---- gate GEMM ----
    gemm_z_kernel<<<dim3(LLEN / (16 * MTILE), DINNER / 16, BATCH), 32, 0, stream>>>(
        xc16, qp16, w_gate16, gate_b, zbuf);

    // ---- chunked selective scan (delta recomputed inline) ----
    scan1_kernel<<<dim3(NCHUNK, BATCH), DINNER, 0, stream>>>(
        dtbuf, dt_w, dt_b, xc, Bsbuf, A_log, cP, cH);
    scan2_kernel<<<dim3(BATCH), DINNER, 0, stream>>>(cP, cH, Hin);
    scan3_kernel<<<dim3(NCHUNK, BATCH), DINNER, 0, stream>>>(
        dtbuf, dt_w, dt_b, xc, Bsbuf, Csbuf, A_log, Hin, zbuf, D_par, yg16);

    // ---- output GEMM (fused epilogue already applied in scan3) ----
    gemm_out_kernel<<<dim3(LLEN / (16 * MTILE), DIMC / 16, BATCH), 32, 0, stream>>>(
        yg16, w_out16, out);

    (void)in_sizes; (void)n_in; (void)out_size; (void)ws_size;
}
